// SinkhornLinear_73031623901731
// MI455X (gfx1250) — matrix operand order repaired and checked by reading, not compiled, not run on hardware
//
#include <hip/hip_runtime.h>

// ---------------------------------------------------------------------------
// SinkhornLinear for MI455X (gfx1250, wave32, WMMA, async-to-LDS)
//   out[b,s,o] = sum_d x[b,s,d] * W[o,d],  W = sinkhorn(weight)
//   x: [8,4096,1024] f32   weight: [1024,1024] f32   out: [8,4096,1024] f32
// ---------------------------------------------------------------------------

#define SIZE 1024
#define M_TOTAL (8 * 4096)   // 32768
#define INV_TEMP 8.0f        // 1 / 0.125

typedef __attribute__((ext_vector_type(16))) __bf16 v16bf;
typedef __attribute__((ext_vector_type(8)))  float  v8f;

union FragBF16 {
    uint4 u[2];
    v16bf v;
};

__device__ __forceinline__ unsigned short f2bf(float f) {
    unsigned int u = __float_as_uint(f);
    unsigned int r = u + 0x7FFFu + ((u >> 16) & 1u);   // round-to-nearest-even
    return (unsigned short)(r >> 16);
}

// CDNA5 async DMA: LDS[lds_off + lane] = MEM[gsrc + lane], 16B per lane,
// tracked by ASYNCcnt. Generic LDS pointer low 32 bits == LDS byte offset.
__device__ __forceinline__ void async_copy_b128(const void* gsrc, void* ldst) {
    unsigned lds_off = (unsigned)(size_t)ldst;
    asm volatile("global_load_async_to_lds_b128 %0, %1, off"
                 :: "v"(lds_off), "v"(gsrc) : "memory");
}

__device__ __forceinline__ void async_wait_all() {
    asm volatile("s_wait_asynccnt 0x0" ::: "memory");
}

// ---------------------------------------------------------------------------
// Sinkhorn kernels (tiny: 4 MB state, L2-resident)
// ---------------------------------------------------------------------------

__global__ void sk_init_kernel(const float* __restrict__ weight,
                               float* __restrict__ logP) {
    int i = blockIdx.x * 256 + threadIdx.x;
    logP[i] = weight[i] * INV_TEMP;
}

// blocks 0..1023   : row logsumexp (one block per row)
// blocks 1024..1055: column partial logsumexp (4 col-groups x 8 row-chunks)
__global__ void sk_lse_kernel(const float* __restrict__ logP,
                              float* __restrict__ rsum,
                              float* __restrict__ pm,
                              float* __restrict__ ps) {
    int t = threadIdx.x;
    if (blockIdx.x < SIZE) {
        int row = blockIdx.x;
        const float* p = logP + (size_t)row * SIZE;
        float m = -INFINITY, s = 0.0f;
        #pragma unroll
        for (int j = 0; j < 4; ++j) {
            float v = p[t + 256 * j];
            float mn = fmaxf(m, v);
            s = s * __expf(m - mn) + __expf(v - mn);
            m = mn;
        }
        __shared__ float sm[256], ss[256];
        sm[t] = m; ss[t] = s;
        __syncthreads();
        for (int off = 128; off > 0; off >>= 1) {
            if (t < off) {
                float m2 = sm[t + off], s2 = ss[t + off];
                float mn = fmaxf(sm[t], m2);
                ss[t] = ss[t] * __expf(sm[t] - mn) + s2 * __expf(m2 - mn);
                sm[t] = mn;
            }
            __syncthreads();
        }
        if (t == 0) rsum[row] = sm[0] + __logf(ss[0]);
    } else {
        int b  = blockIdx.x - SIZE;   // 0..31
        int cg = b >> 3;              // column group 0..3
        int rc = b & 7;               // row chunk  0..7
        int col = cg * 256 + t;
        const float* p = logP + (size_t)rc * 128 * SIZE + col;
        float m = -INFINITY, s = 0.0f;
        for (int r = 0; r < 128; ++r) {
            float v = p[(size_t)r * SIZE];
            float mn = fmaxf(m, v);
            s = s * __expf(m - mn) + __expf(v - mn);
            m = mn;
        }
        pm[rc * SIZE + col] = m;
        ps[rc * SIZE + col] = s;
    }
}

__global__ void sk_combine_kernel(const float* __restrict__ pm,
                                  const float* __restrict__ ps,
                                  float* __restrict__ csum) {
    int col = blockIdx.x * 256 + threadIdx.x;
    float m = -INFINITY, s = 0.0f;
    #pragma unroll
    for (int rc = 0; rc < 8; ++rc) {
        float m2 = pm[rc * SIZE + col], s2 = ps[rc * SIZE + col];
        float mn = fmaxf(m, m2);
        s = s * __expf(m - mn) + s2 * __expf(m2 - mn);
        m = mn;
    }
    csum[col] = m + __logf(s);
}

__global__ void sk_update_kernel(float* __restrict__ logP,
                                 const float* __restrict__ rsum,
                                 const float* __restrict__ csum,
                                 unsigned short* __restrict__ Wb,
                                 int finalize) {
    int i = blockIdx.x * 256 + threadIdx.x;
    float v = logP[i] - 0.5f * (rsum[i >> 10] + csum[i & (SIZE - 1)]);
    if (finalize) {
        Wb[i] = f2bf(__expf(v));   // bf16 doubly-stochastic W, row-major [o][d]
    } else {
        logP[i] = v;
    }
}

// ---------------------------------------------------------------------------
// GEMM: out = x @ W^T via v_wmma_f32_16x16x32_bf16
//   block = 256 thr (8 waves); block tile 128M x 128N; wave tile 64M x 32N
//   double-buffered LDS; W tile staged with global_load_async_to_lds_b128
// ---------------------------------------------------------------------------

__global__ void __launch_bounds__(256)
gemm_bf16_wmma_kernel(const float* __restrict__ x,
                      const unsigned short* __restrict__ Wb,
                      float* __restrict__ out) {
    __shared__ __align__(16) unsigned short lA[2][128 * 32];  // x tile, bf16
    __shared__ __align__(16) unsigned short lB[2][128 * 32];  // W tile, bf16

    const int t    = threadIdx.x;
    const int lane = t & 31;
    const int w    = t >> 5;
    const int mw   = w >> 2;        // 0..1  (wave row in block)
    const int nw   = w & 3;         // 0..3  (wave col in block)
    const int m0   = blockIdx.y * 128;
    const int n0   = blockIdx.x * 128;
    const int half = lane >> 4;     // K-half select per WMMA lane layout
    const int l15  = lane & 15;

    // per-thread staging coordinates (fixed across K-steps)
    const int aRow = t >> 3, aKq = t & 7;       // x tile: 8 float4 per row
    const int bRow = t >> 2, bKq = t & 3;       // W tile: 4 b128 per row

    v8f acc[4][2] = {};

    // ---- stage K-tile `k0` into buffer `buf` ------------------------------
    auto stage = [&](int buf, int k0) {
        // A: x f32 -> bf16 (through VGPRs, conversion required)
        #pragma unroll
        for (int j = 0; j < 4; ++j) {
            int row = aRow + j * 32;
            float4 f = *(const float4*)(x + (size_t)(m0 + row) * SIZE + k0 + aKq * 4);
            unsigned int lo = (unsigned int)f2bf(f.x) | ((unsigned int)f2bf(f.y) << 16);
            unsigned int hi = (unsigned int)f2bf(f.z) | ((unsigned int)f2bf(f.w) << 16);
            *(uint2*)&lA[buf][row * 32 + aKq * 4] = make_uint2(lo, hi);
        }
        // B: bf16 W tile, direct async DMA to LDS (no VGPR round-trip)
        #pragma unroll
        for (int j = 0; j < 2; ++j) {
            int row = bRow + j * 64;
            async_copy_b128(Wb + (size_t)(n0 + row) * SIZE + k0 + bKq * 8,
                            &lB[buf][row * 32 + bKq * 8]);
        }
    };

    stage(0, 0);
    async_wait_all();
    __syncthreads();

    for (int kt = 0; kt < SIZE / 32; ++kt) {
        const int cur = kt & 1;
        if (kt + 1 < SIZE / 32) stage(cur ^ 1, (kt + 1) * 32);

        // --- B fragments: lane<16 -> col=lane,K 0..15; lane>=16 -> K 16..31
        FragBF16 bfr[2];
        #pragma unroll
        for (int nt = 0; nt < 2; ++nt) {
            int ccol = nw * 32 + nt * 16 + l15;
            const uint4* pB = (const uint4*)&lB[cur][ccol * 32 + half * 16];
            bfr[nt].u[0] = pB[0];
            bfr[nt].u[1] = pB[1];
        }
        // --- A fragments + WMMA -------------------------------------------
        #pragma unroll
        for (int mt = 0; mt < 4; ++mt) {
            int r = mw * 64 + mt * 16 + l15;
            FragBF16 af;
            af.u[0] = *(const uint4*)&lA[cur][r * 32 + half * 16];      // K 0-7 / 16-23
            af.u[1] = *(const uint4*)&lA[cur][r * 32 + half * 16 + 8];  // K 8-15 / 24-31
            #pragma unroll
            for (int nt = 0; nt < 2; ++nt) {
                acc[mt][nt] = __builtin_amdgcn_wmma_f32_16x16x32_bf16(
                    false, af.v, false, bfr[nt].v, (short)0, acc[mt][nt],
                    false, false);
            }
        }

        async_wait_all();     // our async stores to LDS (next buffer) landed
        __syncthreads();      // everyone done reading `cur` / staging `cur^1`
    }

    // --- epilogue: C layout (VGPR v: M=v (+8 hi lanes), N=lane) -> global.
    // Non-temporal: `out` is write-once, keep the 192MB L2 for x re-reads.
    #pragma unroll
    for (int mt = 0; mt < 4; ++mt) {
        #pragma unroll
        for (int nt = 0; nt < 2; ++nt) {
            int col   = n0 + nw * 32 + nt * 16 + l15;
            int rbase = m0 + mw * 64 + mt * 16 + (half << 3);
            float* po = out + (size_t)rbase * SIZE + col;
            #pragma unroll
            for (int v = 0; v < 8; ++v) {
                __builtin_nontemporal_store(acc[mt][nt][v], po + (size_t)v * SIZE);
            }
        }
    }
}

// ---------------------------------------------------------------------------
// Launch
// ---------------------------------------------------------------------------

extern "C" void kernel_launch(void* const* d_in, const int* in_sizes, int n_in,
                              void* d_out, int out_size, void* d_ws, size_t ws_size,
                              hipStream_t stream) {
    const float* x      = (const float*)d_in[0];   // [8,4096,1024]
    const float* weight = (const float*)d_in[1];   // [1024,1024]
    float*       out    = (float*)d_out;

    // workspace layout (~6.2 MB)
    float*          logP = (float*)d_ws;                      // 1M f32
    float*          rsum = logP + (1 << 20);                  // 1024
    float*          csum = rsum + SIZE;                       // 1024
    float*          pm   = csum + SIZE;                       // 8*1024
    float*          ps   = pm + 8 * SIZE;                     // 8*1024
    unsigned short* Wb   = (unsigned short*)(ps + 8 * SIZE);  // 1M bf16

    sk_init_kernel<<<4096, 256, 0, stream>>>(weight, logP);
    for (int it = 0; it < 10; ++it) {
        sk_lse_kernel<<<SIZE + 32, 256, 0, stream>>>(logP, rsum, pm, ps);
        sk_combine_kernel<<<4, 256, 0, stream>>>(pm, ps, csum);
        sk_update_kernel<<<4096, 256, 0, stream>>>(logP, rsum, csum, Wb,
                                                   it == 9 ? 1 : 0);
    }

    dim3 grid(SIZE / 128, M_TOTAL / 128);  // (8, 256); N fastest for L2 reuse of x
    gemm_bf16_wmma_kernel<<<grid, 256, 0, stream>>>(x, Wb, out);
}
